// SelfAttention_56599079026811
// MI455X (gfx1250) — matrix operand order repaired and checked by reading, not compiled, not run on hardware
//
#include <hip/hip_runtime.h>

typedef __attribute__((ext_vector_type(16))) _Float16 v16h;
typedef __attribute__((ext_vector_type(8)))  _Float16 v8h;
typedef __attribute__((ext_vector_type(4)))  _Float16 v4h;
typedef __attribute__((ext_vector_type(8)))  float    v8f;
typedef __attribute__((ext_vector_type(8)))  unsigned vu8;
typedef __attribute__((ext_vector_type(4)))  unsigned v4u;
typedef __attribute__((ext_vector_type(8)))  unsigned v8u;

#define D_MODEL 1024
#define NHEAD   16
#define HDIM    64
#define SEQ     2048
#define BATCH   4
#define M_TOK   (BATCH * SEQ)   // 8192
#define N_QKV   (3 * D_MODEL)   // 3072
#define LDS_STR 72              // 64 + 8 halves pad: 144B rows, 16B aligned, conflict-free

// Tile geometry for the TDM-staged GEMMs
#define TILE_HALVES  (128 * LDS_STR)       // 9216 halves per 128x64 tile
#define BUF_HALVES   (2 * TILE_HALVES)     // A+B per buffer
#define SLICE_HALVES (16 * LDS_STR)        // one wave's 16-row slice

// ---------------------------------------------------------------------------
// WMMA wrapper: D = A(16x32 f16) x B(32x16 f16) + C(16x16 f32)
// ---------------------------------------------------------------------------
static __device__ __forceinline__ v8f wmma_f16(v16h a, v16h b, v8f c) {
  return __builtin_amdgcn_wmma_f32_16x16x32_f16(false, a, false, b, (short)0, c,
                                                false, false);
}

// A-fragment (M=16, K=32): lane<16 holds K 0..15, lane>=16 holds K 16..31
// (contiguous per half-wave) of row M=lane&15.  Two 16B loads.
static __device__ __forceinline__ v16h load_fragA(const _Float16* base, int ld,
                                                  int row, int col, int lane) {
  int r = row + (lane & 15);
  int c = col + ((lane >> 4) << 4);
  const _Float16* p = base + (long)r * ld + c;
  v8h lo = *(const v8h*)p;
  v8h hi = *(const v8h*)(p + 8);
  v16h f;
#pragma unroll
  for (int i = 0; i < 8; ++i) { f[i] = lo[i]; f[i + 8] = hi[i]; }
  return f;
}

// B-fragment (K=32, N=16): lane<16 holds K 0..15 of column N=lane, lane>=16
// holds K 16..31 (contiguous per half-wave).  One contiguous 32B read.
static __device__ __forceinline__ v16h load_fragB(const _Float16* base, int ld,
                                                  int nrow, int col, int lane) {
  int n = nrow + (lane & 15);
  int c = col + ((lane >> 4) << 4);
  const _Float16* p = base + (long)n * ld + c;
  v8h lo = *(const v8h*)p;
  v8h hi = *(const v8h*)(p + 8);
  v16h f;
#pragma unroll
  for (int i = 0; i < 8; ++i) { f[i] = lo[i]; f[i + 8] = hi[i]; }
  return f;
}

static __device__ __forceinline__ unsigned pack_h2(float a, float b) {
  unsigned short ha = __builtin_bit_cast(unsigned short, (_Float16)a);
  unsigned short hb = __builtin_bit_cast(unsigned short, (_Float16)b);
  return (unsigned)ha | ((unsigned)hb << 16);
}

// ---------------------------------------------------------------------------
// Tensor Data Mover: async 2D f16 tile (tile_cols x tile_rows) from global to
// LDS, with hardware row padding 32 data DWORDs + 4 pad DWORDs so the LDS row
// stride is 144 B == LDS_STR halves.  D# per CDNA5 ISA.
//
// lds_dst is the generic pointer to the destination slice inside the kernel's
// __shared__ array:
//  * its low 32 bits are the LDS byte offset (generic LDS addr = aperture |
//    offset), used for D#.lds_addr;
//  * it is also passed as an asm operand so the shared array ESCAPES into the
//    asm -- without this, alias analysis proves the asm cannot write smem and
//    deletes the subsequent ds_load fragment reads (observed in round 2).
// ---------------------------------------------------------------------------
static __device__ __forceinline__ void tdm_load_tile_f16(
    _Float16* lds_dst, const _Float16* gsrc, unsigned tensor_rows,
    unsigned row_elems, unsigned tile_cols, unsigned tile_rows) {
  unsigned long ga = (unsigned long)gsrc;
  unsigned lds_off = (unsigned)(unsigned long)lds_dst;  // LDS byte offset
  v4u g0;
  g0[0] = 1u;                                   // count=1, user mode, no gather
  g0[1] = lds_off;                               // lds_addr (bytes)
  g0[2] = (unsigned)(ga & 0xffffffffu);          // global_addr[31:0]
  g0[3] = (unsigned)((ga >> 32) & 0x1ffffffu) |  // global_addr[56:32]
          (2u << 30);                            // type = 2 (image)
  v8u g1;
  g1[0] = (1u << 16)      // data_size = 1 -> 2 bytes
        | (1u << 20)      // pad_enable
        | (4u << 22)      // pad_interval: 2^(4+1)=32 DWORDs (=128B data row)
        | (3u << 25);     // pad_amount: 3+1 = 4 DWORDs (=16B pad)
  g1[1] = (row_elems & 0xffffu) << 16;                       // tensor_dim0 lo16
  g1[2] = ((row_elems >> 16) & 0xffffu) |                    // tensor_dim0 hi16
          ((tensor_rows & 0xffffu) << 16);                   // tensor_dim1 lo16
  g1[3] = ((tensor_rows >> 16) & 0xffffu) |                  // tensor_dim1 hi16
          (tile_cols << 16);                                 // tile_dim0
  g1[4] = tile_rows;                                         // tile_dim1 (dim2=0)
  g1[5] = row_elems;                                         // dim0_stride lo32
  g1[6] = 0u;
  g1[7] = 0u;
  // %2 is not referenced in the template: it exists to make lds_dst escape.
  asm volatile("tensor_load_to_lds %0, %1"
               :
               : "s"(g0), "s"(g1), "v"(lds_dst)
               : "memory");
}

static __device__ __forceinline__ void tdm_wait0() {
  __builtin_amdgcn_s_wait_tensorcnt(0);
}

// ---------------------------------------------------------------------------
// f32 -> f16 conversion (vectorized, bandwidth bound)
// ---------------------------------------------------------------------------
__global__ void cvt_f32_to_f16(const float* __restrict__ in,
                               _Float16* __restrict__ out, int n4) {
  int i = blockIdx.x * blockDim.x + threadIdx.x;
  if (i >= n4) return;
  float4 v = ((const float4*)in)[i];
  v4h o = {(_Float16)v.x, (_Float16)v.y, (_Float16)v.z, (_Float16)v.w};
  ((v4h*)out)[i] = o;
}

// ---------------------------------------------------------------------------
// QKV GEMM: [8192,1024]f16 x [3072,1024]f16^T, TDM-staged double-buffered LDS.
// Block: 256 thr (8 waves), tile 128Mx128N, K-step 64.  Each wave TDM-loads
// its own 16-row slice of the A and B tiles for the NEXT K-step while the
// WMMA pipeline consumes the current buffer.
// Epilogue scatters Q/K row-major per head, V transposed [B,H,d,S] + bias.
// ---------------------------------------------------------------------------
__global__ __launch_bounds__(256) void qkv_gemm_kernel(
    const _Float16* __restrict__ A, const _Float16* __restrict__ W,
    const float* __restrict__ bias, _Float16* __restrict__ qb,
    _Float16* __restrict__ kb, _Float16* __restrict__ vtb) {
  __shared__ _Float16 smem[2 * BUF_HALVES];  // [buf][A|B][...]: 72 KB
  const int tid = threadIdx.x;
  const int lane = tid & 31;
  const int wid = tid >> 5;
  const int uwid = __builtin_amdgcn_readfirstlane(wid);  // provably scalar
  const int wm = wid & 3;   // 32-row strip
  const int wn = wid >> 2;  // 64-col strip
  const int m0 = blockIdx.x * 128;
  const int n0 = blockIdx.y * 128;

  const v8f vz = {0.f, 0.f, 0.f, 0.f, 0.f, 0.f, 0.f, 0.f};
  v8f acc[2][4];
#pragma unroll
  for (int f = 0; f < 2; ++f)
#pragma unroll
    for (int g = 0; g < 4; ++g) acc[f][g] = vz;

  const _Float16* gA = A + (long)(m0 + uwid * 16) * D_MODEL;
  const _Float16* gB = W + (long)(n0 + uwid * 16) * D_MODEL;
  _Float16* sliceA = smem + uwid * SLICE_HALVES;                // buf0 A slice
  _Float16* sliceB = smem + TILE_HALVES + uwid * SLICE_HALVES;  // buf0 B slice

  // Prefetch tile k0=0 into buffer 0
  tdm_load_tile_f16(sliceA, gA, M_TOK, D_MODEL, 64, 16);
  tdm_load_tile_f16(sliceB, gB, N_QKV, D_MODEL, 64, 16);
  tdm_wait0();
  __syncthreads();

  int cur = 0;
  for (int k0 = 0; k0 < D_MODEL; k0 += 64) {
    const int nxt = cur ^ 1;
    if (k0 + 64 < D_MODEL) {  // prefetch next K-step into the other buffer
      tdm_load_tile_f16(sliceA + nxt * BUF_HALVES, gA + k0 + 64, M_TOK,
                        D_MODEL, 64, 16);
      tdm_load_tile_f16(sliceB + nxt * BUF_HALVES, gB + k0 + 64, N_QKV,
                        D_MODEL, 64, 16);
    }
    const _Float16* As = smem + cur * BUF_HALVES;
    const _Float16* Bs = As + TILE_HALVES;
#pragma unroll
    for (int kk = 0; kk < 64; kk += 32) {
      v16h af[2], bf[4];
#pragma unroll
      for (int f = 0; f < 2; ++f)
        af[f] = load_fragA(As, LDS_STR, wm * 32 + f * 16, kk, lane);
#pragma unroll
      for (int g = 0; g < 4; ++g)
        bf[g] = load_fragB(Bs, LDS_STR, wn * 64 + g * 16, kk, lane);
#pragma unroll
      for (int f = 0; f < 2; ++f)
#pragma unroll
        for (int g = 0; g < 4; ++g) acc[f][g] = wmma_f16(af[f], bf[g], acc[f][g]);
    }
    tdm_wait0();      // own slice of next tile landed
    __syncthreads();  // all slices landed; all reads of cur done
    cur = nxt;
  }

  // Epilogue: column n -> head h = n/192, which = (n%192)/64, dim = n%64
#pragma unroll
  for (int f = 0; f < 2; ++f) {
    const int mb = m0 + wm * 32 + f * 16 + ((lane >> 4) << 3);
#pragma unroll
    for (int g = 0; g < 4; ++g) {
      const int n = n0 + wn * 64 + g * 16 + (lane & 15);
      const float bv = bias[n];
      const int h = n / 192;
      const int rem = n - h * 192;
      const int which = rem >> 6;
      const int dc = rem & 63;
#pragma unroll
      for (int r = 0; r < 8; ++r) {
        const int m = mb + r;
        const int b = m >> 11;
        const int s = m & (SEQ - 1);
        const _Float16 hv = (_Float16)(acc[f][g][r] + bv);
        const long bh = (long)(b * NHEAD + h);
        if (which == 0)      qb[(bh * SEQ + s) * HDIM + dc] = hv;
        else if (which == 1) kb[(bh * SEQ + s) * HDIM + dc] = hv;
        else                 vtb[(bh * HDIM + dc) * SEQ + s] = hv;
      }
    }
  }
}

// ---------------------------------------------------------------------------
// Flash attention, transposed formulation:  S^T = K Q^T ; O^T = V^T P^T.
// Wave owns 32 q rows; block = 8 waves = 256 q rows; grid (8, B*H).
// Softmax in exp2 domain (Q pre-scaled by 1/sqrt(d) * log2 e).
// ---------------------------------------------------------------------------
__global__ __launch_bounds__(256) void attn_kernel(
    const _Float16* __restrict__ qb, const _Float16* __restrict__ kb,
    const _Float16* __restrict__ vtb, _Float16* __restrict__ ob) {
  const int tid = threadIdx.x;
  const int lane = tid & 31;
  const int wid = tid >> 5;
  const int bh = blockIdx.y;
  const int q0 = blockIdx.x * 256 + wid * 32;

  const _Float16* Q  = qb  + (long)bh * SEQ * HDIM;
  const _Float16* K  = kb  + (long)bh * SEQ * HDIM;
  const _Float16* Vt = vtb + (long)bh * HDIM * SEQ;

  // Q^T as B-fragments (lane = query column), pre-scaled.
  const _Float16 sc = (_Float16)(0.125f * 1.44269504088896f);
  v16h bq[2][2];
#pragma unroll
  for (int qf = 0; qf < 2; ++qf)
#pragma unroll
    for (int kd = 0; kd < 2; ++kd) {
      v16h t = load_fragB(Q, HDIM, q0 + qf * 16, kd * 32, lane);
      bq[qf][kd] = t * sc;
    }

  const v8f vz = {0.f, 0.f, 0.f, 0.f, 0.f, 0.f, 0.f, 0.f};
  v8f vacc[4][2];
#pragma unroll
  for (int df = 0; df < 4; ++df)
#pragma unroll
    for (int qf = 0; qf < 2; ++qf) vacc[df][qf] = vz;
  float m_run[2] = {-__builtin_inff(), -__builtin_inff()};
  float l_run[2] = {0.f, 0.f};

  for (int kc = 0; kc < SEQ; kc += 32) {
    v16h ka[2][2];
#pragma unroll
    for (int ks = 0; ks < 2; ++ks)
#pragma unroll
      for (int kd = 0; kd < 2; ++kd)
        ka[ks][kd] = load_fragA(K, HDIM, kc + ks * 16, kd * 32, lane);

    // S^T chunk: [32 keys x 16 q] per qf (lane holds one q column)
    v8f st[2][2];
#pragma unroll
    for (int ks = 0; ks < 2; ++ks)
#pragma unroll
      for (int qf = 0; qf < 2; ++qf) {
        v8f s = vz;
        s = wmma_f16(ka[ks][0], bq[qf][0], s);
        s = wmma_f16(ka[ks][1], bq[qf][1], s);
        st[ks][qf] = s;
      }

    v16h pb[2];
#pragma unroll
    for (int qf = 0; qf < 2; ++qf) {
      // per-q max: 16 in-lane values + one xor-16 partner exchange
      float cmax = -__builtin_inff();
#pragma unroll
      for (int ks = 0; ks < 2; ++ks)
#pragma unroll
        for (int r = 0; r < 8; ++r) cmax = fmaxf(cmax, st[ks][qf][r]);
      cmax = fmaxf(cmax, __shfl_xor(cmax, 16, 32));
      const float m_new = fmaxf(m_run[qf], cmax);
      const float corr = exp2f(m_run[qf] - m_new);
      l_run[qf] *= corr;
#pragma unroll
      for (int df = 0; df < 4; ++df) vacc[df][qf] *= corr;

      float psum = 0.f;
#pragma unroll
      for (int ks = 0; ks < 2; ++ks)
#pragma unroll
        for (int r = 0; r < 8; ++r) {
          const float p = exp2f(st[ks][qf][r] - m_new);
          st[ks][qf][r] = p;
          psum += p;
        }
      psum += __shfl_xor(psum, 16, 32);
      l_run[qf] += psum;
      m_run[qf] = m_new;

      // Re-pack P^T (C layout) into B-fragment via 8 word shuffles:
      // lane<16 needs keys 0..15 (own ks0 rows + partner ks0 rows),
      // lane>=16 needs keys 16..31 (partner ks1 rows + own ks1 rows).
      unsigned u0[4], u1[4];
#pragma unroll
      for (int i = 0; i < 4; ++i) {
        u0[i] = pack_h2(st[0][qf][2 * i], st[0][qf][2 * i + 1]);
        u1[i] = pack_h2(st[1][qf][2 * i], st[1][qf][2 * i + 1]);
      }
      vu8 pw;
      const bool loHalf = (lane < 16);
#pragma unroll
      for (int i = 0; i < 4; ++i) {
        const unsigned t0 = (unsigned)__shfl_xor((int)u0[i], 16, 32);
        const unsigned t1 = (unsigned)__shfl_xor((int)u1[i], 16, 32);
        pw[i]     = loHalf ? u0[i] : t1;
        pw[4 + i] = loHalf ? t0 : u1[i];
      }
      pb[qf] = __builtin_bit_cast(v16h, pw);
    }

    // O^T += V^T[d x 32keys] x P^T[32keys x 16q]
    v16h va[4];
#pragma unroll
    for (int df = 0; df < 4; ++df)
      va[df] = load_fragA(Vt, SEQ, df * 16, kc, lane);
#pragma unroll
    for (int df = 0; df < 4; ++df)
#pragma unroll
      for (int qf = 0; qf < 2; ++qf)
        vacc[df][qf] = wmma_f16(va[df], pb[qf], vacc[df][qf]);
  }

  // Normalize and store: lane's 8 accum values are 8 consecutive d -> 16B store
  const int b = bh >> 4;
  const int h = bh & (NHEAD - 1);
#pragma unroll
  for (int qf = 0; qf < 2; ++qf) {
    const float inv = 1.0f / l_run[qf];
    const int q = q0 + qf * 16 + (lane & 15);
    const long row = ((long)b * SEQ + q) * (long)D_MODEL;
#pragma unroll
    for (int df = 0; df < 4; ++df) {
      v8h o;
#pragma unroll
      for (int r = 0; r < 8; ++r) o[r] = (_Float16)(vacc[df][qf][r] * inv);
      const int d0 = h * HDIM + df * 16 + ((lane >> 4) << 3);
      *(v8h*)(ob + row + d0) = o;
    }
  }
}

// ---------------------------------------------------------------------------
// Output projection: attn[8192,1024]f16 x w_o[1024,1024]f16^T + b_o -> f32
// Same TDM-staged double-buffered scheme as the QKV GEMM.
// ---------------------------------------------------------------------------
__global__ __launch_bounds__(256) void oproj_gemm_kernel(
    const _Float16* __restrict__ A, const _Float16* __restrict__ W,
    const float* __restrict__ bias, float* __restrict__ out) {
  __shared__ _Float16 smem[2 * BUF_HALVES];
  const int tid = threadIdx.x;
  const int lane = tid & 31;
  const int wid = tid >> 5;
  const int uwid = __builtin_amdgcn_readfirstlane(wid);
  const int wm = wid & 3;
  const int wn = wid >> 2;
  const int m0 = blockIdx.x * 128;
  const int n0 = blockIdx.y * 128;

  const v8f vz = {0.f, 0.f, 0.f, 0.f, 0.f, 0.f, 0.f, 0.f};
  v8f acc[2][4];
#pragma unroll
  for (int f = 0; f < 2; ++f)
#pragma unroll
    for (int g = 0; g < 4; ++g) acc[f][g] = vz;

  const _Float16* gA = A + (long)(m0 + uwid * 16) * D_MODEL;
  const _Float16* gB = W + (long)(n0 + uwid * 16) * D_MODEL;
  _Float16* sliceA = smem + uwid * SLICE_HALVES;
  _Float16* sliceB = smem + TILE_HALVES + uwid * SLICE_HALVES;

  tdm_load_tile_f16(sliceA, gA, M_TOK, D_MODEL, 64, 16);
  tdm_load_tile_f16(sliceB, gB, D_MODEL, D_MODEL, 64, 16);
  tdm_wait0();
  __syncthreads();

  int cur = 0;
  for (int k0 = 0; k0 < D_MODEL; k0 += 64) {
    const int nxt = cur ^ 1;
    if (k0 + 64 < D_MODEL) {
      tdm_load_tile_f16(sliceA + nxt * BUF_HALVES, gA + k0 + 64, M_TOK,
                        D_MODEL, 64, 16);
      tdm_load_tile_f16(sliceB + nxt * BUF_HALVES, gB + k0 + 64, D_MODEL,
                        D_MODEL, 64, 16);
    }
    const _Float16* As = smem + cur * BUF_HALVES;
    const _Float16* Bs = As + TILE_HALVES;
#pragma unroll
    for (int kk = 0; kk < 64; kk += 32) {
      v16h af[2], bf[4];
#pragma unroll
      for (int f = 0; f < 2; ++f)
        af[f] = load_fragA(As, LDS_STR, wm * 32 + f * 16, kk, lane);
#pragma unroll
      for (int g = 0; g < 4; ++g)
        bf[g] = load_fragB(Bs, LDS_STR, wn * 64 + g * 16, kk, lane);
#pragma unroll
      for (int f = 0; f < 2; ++f)
#pragma unroll
        for (int g = 0; g < 4; ++g) acc[f][g] = wmma_f16(af[f], bf[g], acc[f][g]);
    }
    tdm_wait0();
    __syncthreads();
    cur = nxt;
  }

#pragma unroll
  for (int f = 0; f < 2; ++f) {
    const int mb = m0 + wm * 32 + f * 16 + ((lane >> 4) << 3);
#pragma unroll
    for (int g = 0; g < 4; ++g) {
      const int n = n0 + wn * 64 + g * 16 + (lane & 15);
      const float bv = bias[n];
#pragma unroll
      for (int r = 0; r < 8; ++r)
        out[(long)(mb + r) * D_MODEL + n] = acc[f][g][r] + bv;
    }
  }
}

// ---------------------------------------------------------------------------
// Launcher.  Workspace layout (needs ~92 MB):
//   xh[8192x1024]h, wqkvh[3072x1024]h, woh[1024x1024]h,
//   Q,K [B,H,S,d]h, Vt [B,H,d,S]h, attn[8192x1024]h
// ---------------------------------------------------------------------------
extern "C" void kernel_launch(void* const* d_in, const int* in_sizes, int n_in,
                              void* d_out, int out_size, void* d_ws,
                              size_t ws_size, hipStream_t stream) {
  (void)in_sizes; (void)n_in; (void)out_size; (void)ws_size;
  const float* x      = (const float*)d_in[0];
  const float* w_qkv  = (const float*)d_in[1];
  const float* b_qkv  = (const float*)d_in[2];
  const float* w_o    = (const float*)d_in[3];
  const float* b_o    = (const float*)d_in[4];
  float* out = (float*)d_out;

  char* ws = (char*)d_ws;
  _Float16* xh    = (_Float16*)ws;               ws += (size_t)M_TOK * D_MODEL * 2;
  _Float16* wqkvh = (_Float16*)ws;               ws += (size_t)N_QKV * D_MODEL * 2;
  _Float16* woh   = (_Float16*)ws;               ws += (size_t)D_MODEL * D_MODEL * 2;
  _Float16* qb    = (_Float16*)ws;               ws += (size_t)M_TOK * D_MODEL * 2;
  _Float16* kb    = (_Float16*)ws;               ws += (size_t)M_TOK * D_MODEL * 2;
  _Float16* vtb   = (_Float16*)ws;               ws += (size_t)M_TOK * D_MODEL * 2;
  _Float16* attn  = (_Float16*)ws;               ws += (size_t)M_TOK * D_MODEL * 2;

  const int n4x = M_TOK * D_MODEL / 4;
  const int n4q = N_QKV * D_MODEL / 4;
  const int n4o = D_MODEL * D_MODEL / 4;
  cvt_f32_to_f16<<<(n4x + 255) / 256, 256, 0, stream>>>(x, xh, n4x);
  cvt_f32_to_f16<<<(n4q + 255) / 256, 256, 0, stream>>>(w_qkv, wqkvh, n4q);
  cvt_f32_to_f16<<<(n4o + 255) / 256, 256, 0, stream>>>(w_o, woh, n4o);

  qkv_gemm_kernel<<<dim3(M_TOK / 128, N_QKV / 128), 256, 0, stream>>>(
      xh, wqkvh, b_qkv, qb, kb, vtb);

  attn_kernel<<<dim3(SEQ / 256, BATCH * NHEAD), 256, 0, stream>>>(qb, kb, vtb,
                                                                  attn);

  oproj_gemm_kernel<<<dim3(M_TOK / 128, D_MODEL / 128), 256, 0, stream>>>(
      attn, woh, b_o, out);
}
